// BERT_21174188769851
// MI455X (gfx1250) — matrix operand order repaired and checked by reading, not compiled, not run on hardware
//
#include <hip/hip_runtime.h>
#include <hip/hip_bf16.h>

// ---------------------------------------------------------------------------
// GPT-1 style transformer forward for MI455X (gfx1250, wave32, WMMA).
// All matmuls via v_wmma_f32_16x16x32_bf16 (fp32 accumulate).
// Weights are pre-transposed+converted to bf16 [N,K] once per layer so the
// GEMM hot loop does only b128 LDS copies and WMMA.
// ---------------------------------------------------------------------------

typedef __attribute__((ext_vector_type(16))) __bf16 v16bf;
typedef __attribute__((ext_vector_type(8)))  float  v8f;

#define NB   8
#define NS   512
#define ND   768
#define NH   12
#define NDH  64
#define NDF  3072
#define NL   12
#define NV   40478
#define NM   (NB * NS)        // 4096 rows
#define TD   (3 * ND)         // 2304

__device__ __forceinline__ unsigned short f32_to_bf16(float f) {
    unsigned u = __float_as_uint(f);
    unsigned r = u + 0x7FFFu + ((u >> 16) & 1u);   // round-to-nearest-even
    return (unsigned short)(r >> 16);
}

// two floats -> packed bf16x2 (lo in bits[15:0] = lower LDS address)
__device__ __forceinline__ unsigned pack_bf16x2(float lo, float hi) {
    return (unsigned)f32_to_bf16(lo) | ((unsigned)f32_to_bf16(hi) << 16);
}

__device__ __forceinline__ float gelu_f(float x) {
    const float c = 0.7978845608028654f;           // sqrt(2/pi)
    float x3 = x * x * x;
    return 0.5f * x * (1.0f + tanhf(c * (x + 0.044715f * x3)));
}

union FragU {
    struct { uint4 lo, hi; } q;
    v16bf v;
};

// A fragment (16x32 bf16, M=lane&15): lanes 0-15 hold K={kb..kb+7, kb+16..kb+23}
// with kb=0; lanes 16-31 the same with kb=8.  (ISA 7.12.2 16-bit A layout)
__device__ __forceinline__ v16bf load_frag_A(const unsigned short* base, int ld, int lane) {
    int row = lane & 15;
    int kb  = (lane < 16) ? 0 : 8;
    const unsigned short* p = base + row * ld + kb;
    FragU u;
    u.q.lo = *reinterpret_cast<const uint4*>(p);
    u.q.hi = *reinterpret_cast<const uint4*>(p + 16);
    return u.v;
}

// B fragment (32x16 bf16, N=lane&15): lanes 0-15 hold K={0..7,16..23},
// lanes 16-31 K={8..15,24..31}; LDS stores B column-major [N][K].
__device__ __forceinline__ v16bf load_frag_B(const unsigned short* base, int ld, int lane) {
    int col = lane & 15;
    int kb  = (lane < 16) ? 0 : 8;
    const unsigned short* p = base + col * ld + kb;
    FragU u;
    u.q.lo = *reinterpret_cast<const uint4*>(p);
    u.q.hi = *reinterpret_cast<const uint4*>(p + 16);
    return u.v;
}

__device__ __forceinline__ v8f wmma_bf16(v16bf a, v16bf b, v8f c) {
    return __builtin_amdgcn_wmma_f32_16x16x32_bf16(false, a, false, b, (short)0, c, false, false);
}

// ---------------------------------------------------------------------------
// Weight transpose + convert: src f32 [K,N] -> dst bf16 [N,K].  64x64 tiles.
// ---------------------------------------------------------------------------
#define TLD 72

__global__ __launch_bounds__(256) void wtrans_kernel(
    const float* __restrict__ src, unsigned short* __restrict__ dst,
    int K, int N) {
    __shared__ unsigned short tileL[64 * TLD];    // [n_local][k_local]
    const int t     = threadIdx.x;
    const int nBase = blockIdx.x * 64;
    const int kBase = blockIdx.y * 64;

    // load 64x64 f32 coalesced along N; pack bf16 pairs along K into b32 stores
#pragma unroll
    for (int i = 0; i < 2; ++i) {
        int pc   = t + i * 256;            // 0..511 pair chunks
        int kp   = pc >> 4;                // k-row pair 0..31
        int col4 = (pc & 15) << 2;         // n_local 0..60
        const float4 w0 = *reinterpret_cast<const float4*>(
            src + (size_t)(kBase + 2 * kp) * N + nBase + col4);
        const float4 w1 = *reinterpret_cast<const float4*>(
            src + (size_t)(kBase + 2 * kp + 1) * N + nBase + col4);
        *reinterpret_cast<unsigned*>(&tileL[(col4 + 0) * TLD + 2 * kp]) = pack_bf16x2(w0.x, w1.x);
        *reinterpret_cast<unsigned*>(&tileL[(col4 + 1) * TLD + 2 * kp]) = pack_bf16x2(w0.y, w1.y);
        *reinterpret_cast<unsigned*>(&tileL[(col4 + 2) * TLD + 2 * kp]) = pack_bf16x2(w0.z, w1.z);
        *reinterpret_cast<unsigned*>(&tileL[(col4 + 3) * TLD + 2 * kp]) = pack_bf16x2(w0.w, w1.w);
    }
    __syncthreads();

    // store bf16 rows (contiguous K) coalesced as b128
#pragma unroll
    for (int i = 0; i < 2; ++i) {
        int oc   = t + i * 256;            // 0..511 (64 rows x 8 chunks)
        int nrow = oc >> 3;
        int k8   = (oc & 7) << 3;
        uint4 d = *reinterpret_cast<const uint4*>(&tileL[nrow * TLD + k8]);
        *reinterpret_cast<uint4*>(dst + (size_t)(nBase + nrow) * K + kBase + k8) = d;
    }
}

// ---------------------------------------------------------------------------
// Embedding: h[b,s,:] = we[token] + we[V+s]; writes f32 + bf16.
// ---------------------------------------------------------------------------
__global__ __launch_bounds__(256) void embed_kernel(
    const int* __restrict__ tokens, const float* __restrict__ we,
    float* __restrict__ X, unsigned short* __restrict__ Xb) {
    int row = blockIdx.x;                 // 0..4095
    int s   = row & (NS - 1);
    int tok = tokens[row];
    size_t tb = (size_t)tok * ND;
    size_t pb = (size_t)(NV + s) * ND;
    size_t ob = (size_t)row * ND;
#pragma unroll
    for (int i = 0; i < 3; ++i) {
        int d = threadIdx.x + i * 256;
        float v = we[tb + d] + we[pb + d];
        X[ob + d]  = v;
        Xb[ob + d] = f32_to_bf16(v);
    }
}

// ---------------------------------------------------------------------------
// GEMM: C[M,N] = act(A_bf16[M,K] @ Wt_bf16[N,K]^T + bias).  Block tile
// 128x128, K-step 32, 8 waves each computing 32x64 via 8 WMMA per step.
// Both A and B staging are pure b128 copies (no conversion).
// ---------------------------------------------------------------------------
#define ALD 40   // padded LDS strides (bf16 elems) -> conflict-free frag loads
#define BLD 40

__global__ __launch_bounds__(256) void gemm_bf16_kernel(
    const unsigned short* __restrict__ A, const unsigned short* __restrict__ Wt,
    const float* __restrict__ bias, float* __restrict__ Cf,
    unsigned short* __restrict__ Cb, int M, int N, int K, int act) {
    __shared__ unsigned short sA[128 * ALD];
    __shared__ unsigned short sB[128 * BLD];

    const int t    = threadIdx.x;
    const int lane = t & 31;
    const int wave = t >> 5;
    const int wm   = wave & 3;    // 4 subtiles of 32 rows
    const int wn   = wave >> 2;   // 2 subtiles of 64 cols
    const int rowBase = blockIdx.y * 128;
    const int colBase = blockIdx.x * 128;

    const int sRow  = t >> 1;     // staging row/col 0..127
    const int sHalf = t & 1;      // 16-elem half of a 32-K row

    v8f acc[2][4];
#pragma unroll
    for (int i = 0; i < 2; ++i)
#pragma unroll
        for (int j = 0; j < 4; ++j)
#pragma unroll
            for (int e = 0; e < 8; ++e) acc[i][j][e] = 0.0f;

    for (int k0 = 0; k0 < K; k0 += 32) {
        // ---- stage A tile (128x32 bf16): 2x uint4 per thread
        {
            const uint4* gp = reinterpret_cast<const uint4*>(
                A + (size_t)(rowBase + sRow) * K + k0 + sHalf * 16);
            uint4 d0 = gp[0];
            uint4 d1 = gp[1];
            uint4* sp = reinterpret_cast<uint4*>(&sA[sRow * ALD + sHalf * 16]);
            sp[0] = d0;
            sp[1] = d1;
        }
        // ---- stage B tile (128 cols x 32 K bf16 from Wt[N,K]): 2x uint4
        {
            const uint4* gp = reinterpret_cast<const uint4*>(
                Wt + (size_t)(colBase + sRow) * K + k0 + sHalf * 16);
            uint4 d0 = gp[0];
            uint4 d1 = gp[1];
            uint4* sp = reinterpret_cast<uint4*>(&sB[sRow * BLD + sHalf * 16]);
            sp[0] = d0;
            sp[1] = d1;
        }
        // prefetch next Wt tile rows into cache (global_prefetch_b8)
        if (k0 + 32 < K)
            __builtin_prefetch(Wt + (size_t)(colBase + sRow) * K + k0 + 32, 0, 1);
        __syncthreads();

        v16bf af[2], bf_[4];
#pragma unroll
        for (int mf = 0; mf < 2; ++mf)
            af[mf] = load_frag_A(sA + (wm * 32 + mf * 16) * ALD, ALD, lane);
#pragma unroll
        for (int nf = 0; nf < 4; ++nf)
            bf_[nf] = load_frag_B(sB + (wn * 64 + nf * 16) * BLD, BLD, lane);
#pragma unroll
        for (int mf = 0; mf < 2; ++mf)
#pragma unroll
            for (int nf = 0; nf < 4; ++nf)
                acc[mf][nf] = wmma_bf16(af[mf], bf_[nf], acc[mf][nf]);
        __syncthreads();
    }

    // ---- epilogue: bias (+ GELU), write f32 / bf16
    const int colLane = lane & 15;
    const int hi8     = (lane < 16) ? 0 : 8;
#pragma unroll
    for (int mf = 0; mf < 2; ++mf) {
        int r0 = rowBase + wm * 32 + mf * 16 + hi8;
#pragma unroll
        for (int nf = 0; nf < 4; ++nf) {
            int col = colBase + wn * 64 + nf * 16 + colLane;
            float bv = bias[col];
#pragma unroll
            for (int r = 0; r < 8; ++r) {
                float v = acc[mf][nf][r] + bv;
                if (act == 1) v = gelu_f(v);
                size_t idx = (size_t)(r0 + r) * N + col;
                if (Cf) Cf[idx] = v;
                if (Cb) Cb[idx] = f32_to_bf16(v);
            }
        }
    }
}

// ---------------------------------------------------------------------------
// Flash attention: one WG per (b, h, 128 q-rows); key blocks of 64.
// QK^T and P@V via WMMA bf16; online softmax on C-layout accumulators.
// Consumes bf16 QKV produced by the QKV GEMM (b128 copy staging).
// ---------------------------------------------------------------------------
#define QLD 72
#define KLD 72
#define VLD 72
#define PLD 72

__global__ __launch_bounds__(256) void attn_kernel(
    const unsigned short* __restrict__ QKV, unsigned short* __restrict__ Ab) {
    __shared__ unsigned short qL[128 * QLD];     // [qrow][d]
    __shared__ unsigned short kL[64 * KLD];      // [key][d]   (B col-major for QK^T)
    __shared__ unsigned short vL[64 * VLD];      // [d][key]   (B col-major for P@V)
    __shared__ unsigned short pL[8 * 16 * PLD];  // per-wave P rows [16][keypad]

    const int t    = threadIdx.x;
    const int lane = t & 31;
    const int wave = t >> 5;
    const int qb   = blockIdx.x;           // 0..3 (128-row q block)
    const int b    = blockIdx.y / NH;
    const int h    = blockIdx.y % NH;
    const int colLane = lane & 15;
    const int hi8     = (lane < 16) ? 0 : 8;

    // stage q block (128 x 64 bf16) -> b128 copies
#pragma unroll
    for (int i = 0; i < 4; ++i) {
        int chunk = t + i * 256;            // 1024 uint4 chunks (8 per row)
        int row   = chunk >> 3;
        int d8    = (chunk & 7) << 3;
        uint4 d = *reinterpret_cast<const uint4*>(
            QKV + (size_t)(b * NS + qb * 128 + row) * TD + h * NDH + d8);
        *reinterpret_cast<uint4*>(&qL[row * QLD + d8]) = d;
    }

    float rowm[8], rowl[8];
    v8f oacc[4];
#pragma unroll
    for (int r = 0; r < 8; ++r) { rowm[r] = -3.0e38f; rowl[r] = 0.0f; }
#pragma unroll
    for (int o = 0; o < 4; ++o)
#pragma unroll
        for (int e = 0; e < 8; ++e) oacc[o][e] = 0.0f;

    const int nkb = 2 * qb + 2;            // key blocks of 64 covering causal range
    for (int kb = 0; kb < nkb; ++kb) {
        // ---- stage k (64x64 bf16 -> [key][d]) via b128 copies
#pragma unroll
        for (int i = 0; i < 2; ++i) {
            int chunk = t + i * 256;        // 512 uint4 chunks
            int row   = chunk >> 3;         // key 0..63
            int d8    = (chunk & 7) << 3;
            uint4 d = *reinterpret_cast<const uint4*>(
                QKV + (size_t)(b * NS + kb * 64 + row) * TD + ND + h * NDH + d8);
            *reinterpret_cast<uint4*>(&kL[row * KLD + d8]) = d;
        }
        // ---- stage v transposed (64x64 bf16 -> [d][key]); two consecutive
        // key rows per thread, 16-bit pair swizzle (v_perm) along key.
        {
            int pc = t;                     // 256 pair chunks (32 pairs x 8)
            int rp = pc >> 3;               // key pair 0..31
            int d8 = (pc & 7) << 3;         // d 0..56
            size_t gb = (size_t)(b * NS + kb * 64 + 2 * rp) * TD + 2 * ND + h * NDH + d8;
            uint4 a = *reinterpret_cast<const uint4*>(QKV + gb);
            uint4 c = *reinterpret_cast<const uint4*>(QKV + gb + TD);
            const unsigned* av = reinterpret_cast<const unsigned*>(&a);
            const unsigned* cv = reinterpret_cast<const unsigned*>(&c);
#pragma unroll
            for (int j2 = 0; j2 < 4; ++j2) {
                unsigned x = av[j2], y = cv[j2];
                unsigned p0 = (x & 0x0000FFFFu) | (y << 16);         // d8+2*j2
                unsigned p1 = (x >> 16)         | (y & 0xFFFF0000u); // d8+2*j2+1
                *reinterpret_cast<unsigned*>(&vL[(d8 + 2 * j2 + 0) * VLD + 2 * rp]) = p0;
                *reinterpret_cast<unsigned*>(&vL[(d8 + 2 * j2 + 1) * VLD + 2 * rp]) = p1;
            }
        }
        __syncthreads();

        // ---- scores: 16 q-rows x 64 keys, K = DH = 64 (two WMMA steps)
        v8f sc[4];
#pragma unroll
        for (int nf = 0; nf < 4; ++nf)
#pragma unroll
            for (int e = 0; e < 8; ++e) sc[nf][e] = 0.0f;
#pragma unroll
        for (int k0 = 0; k0 < 64; k0 += 32) {
            v16bf aq = load_frag_A(qL + (wave * 16) * QLD + k0, QLD, lane);
#pragma unroll
            for (int nf = 0; nf < 4; ++nf) {
                v16bf bk = load_frag_B(kL + (nf * 16) * KLD + k0, KLD, lane);
                sc[nf] = wmma_bf16(aq, bk, sc[nf]);
            }
        }

        // ---- scale + causal mask
        const int qrow0 = qb * 128 + wave * 16 + hi8;
#pragma unroll
        for (int nf = 0; nf < 4; ++nf) {
            int kcol = kb * 64 + nf * 16 + colLane;
#pragma unroll
            for (int r = 0; r < 8; ++r) {
                float s = sc[nf][r] * 0.125f;       // 1/sqrt(64)
                if (kcol > qrow0 + r) s = -1.0e9f;
                sc[nf][r] = s;
            }
        }

        // ---- online softmax (row stats via intra-16-lane xor reductions)
        unsigned short* pw = pL + wave * 16 * PLD;
#pragma unroll
        for (int r = 0; r < 8; ++r) {
            float bm = sc[0][r];
#pragma unroll
            for (int nf = 1; nf < 4; ++nf) bm = fmaxf(bm, sc[nf][r]);
#pragma unroll
            for (int off = 1; off < 16; off <<= 1)
                bm = fmaxf(bm, __shfl_xor(bm, off, 32));
            float nm    = fmaxf(rowm[r], bm);
            float scale = __expf(rowm[r] - nm);
            rowm[r] = nm;
            float rs = 0.0f;
#pragma unroll
            for (int nf = 0; nf < 4; ++nf) {
                float e = __expf(sc[nf][r] - nm);
                sc[nf][r] = e;
                rs += e;
            }
#pragma unroll
            for (int off = 1; off < 16; off <<= 1)
                rs += __shfl_xor(rs, off, 32);
            rowl[r] = rowl[r] * scale + rs;
#pragma unroll
            for (int o = 0; o < 4; ++o) oacc[o][r] *= scale;
        }

        // ---- write P (bf16) into per-wave LDS for A-fragment reload
#pragma unroll
        for (int nf = 0; nf < 4; ++nf)
#pragma unroll
            for (int r = 0; r < 8; ++r)
                pw[(r + hi8) * PLD + nf * 16 + colLane] = f32_to_bf16(sc[nf][r]);
        // same-wave LDS ops are in-order; no cross-wave hazard (private region)

        // ---- out += P @ V  (K = 64 keys, two WMMA steps, 4 output col-frags)
#pragma unroll
        for (int k0 = 0; k0 < 64; k0 += 32) {
            v16bf ap = load_frag_A(pw + k0, PLD, lane);
#pragma unroll
            for (int o = 0; o < 4; ++o) {
                v16bf bv = load_frag_B(vL + (o * 16) * VLD + k0, VLD, lane);
                oacc[o] = wmma_bf16(ap, bv, oacc[o]);
            }
        }
        __syncthreads();   // protect kL/vL for next key block
    }

    // ---- normalize and store attention output (bf16, [B*S, D] at head cols)
#pragma unroll
    for (int o = 0; o < 4; ++o) {
        int col = h * NDH + o * 16 + colLane;
#pragma unroll
        for (int r = 0; r < 8; ++r) {
            int rowg = b * NS + qb * 128 + wave * 16 + hi8 + r;
            Ab[(size_t)rowg * ND + col] = f32_to_bf16(oacc[o][r] / rowl[r]);
        }
    }
}

// ---------------------------------------------------------------------------
// Residual + LayerNorm: Y = g * norm(X + R) + b; writes f32 + bf16.
// ---------------------------------------------------------------------------
__global__ __launch_bounds__(256) void add_ln_kernel(
    const float* __restrict__ X, const float* __restrict__ R,
    const float* __restrict__ g, const float* __restrict__ bt,
    float* __restrict__ Y, unsigned short* __restrict__ Yb) {
    __shared__ float red[256];
    const int t = threadIdx.x;
    const size_t base = (size_t)blockIdx.x * ND;

    float v0 = X[base + t]       + R[base + t];
    float v1 = X[base + t + 256] + R[base + t + 256];
    float v2 = X[base + t + 512] + R[base + t + 512];

    red[t] = v0 + v1 + v2;
    __syncthreads();
    for (int s = 128; s > 0; s >>= 1) {
        if (t < s) red[t] += red[t + s];
        __syncthreads();
    }
    float mean = red[0] * (1.0f / ND);
    __syncthreads();

    float d0 = v0 - mean, d1 = v1 - mean, d2 = v2 - mean;
    red[t] = d0 * d0 + d1 * d1 + d2 * d2;
    __syncthreads();
    for (int s = 128; s > 0; s >>= 1) {
        if (t < s) red[t] += red[t + s];
        __syncthreads();
    }
    float rstd = rsqrtf(red[0] * (1.0f / ND) + 1e-5f);

    float y0 = g[t]       * d0 * rstd + bt[t];
    float y1 = g[t + 256] * d1 * rstd + bt[t + 256];
    float y2 = g[t + 512] * d2 * rstd + bt[t + 512];
    Y[base + t]        = y0;  Yb[base + t]        = f32_to_bf16(y0);
    Y[base + t + 256]  = y1;  Yb[base + t + 256]  = f32_to_bf16(y1);
    Y[base + t + 512]  = y2;  Yb[base + t + 512]  = f32_to_bf16(y2);
}

// ---------------------------------------------------------------------------
// Host orchestration
// ---------------------------------------------------------------------------
extern "C" void kernel_launch(void* const* d_in, const int* in_sizes, int n_in,
                              void* d_out, int out_size, void* d_ws, size_t ws_size,
                              hipStream_t stream) {
    const int*   tokens = (const int*)  d_in[0];
    const float* we     = (const float*)d_in[1];
    const float* wqkv   = (const float*)d_in[2];
    const float* bqkv   = (const float*)d_in[3];
    const float* wproj  = (const float*)d_in[4];
    const float* bproj  = (const float*)d_in[5];
    const float* g1     = (const float*)d_in[6];
    const float* b1     = (const float*)d_in[7];
    const float* wfc    = (const float*)d_in[8];
    const float* bfc    = (const float*)d_in[9];
    const float* wpr    = (const float*)d_in[10];
    const float* bpr    = (const float*)d_in[11];
    const float* g2     = (const float*)d_in[12];
    const float* b2     = (const float*)d_in[13];

    char* ws = (char*)d_ws;
    size_t off = 0;
    auto alloc = [&](size_t bytes) {
        char* p = ws + off;
        off = (off + bytes + 255) & ~(size_t)255;
        return p;
    };
    float*          X    = (float*)         alloc((size_t)NM * ND  * 4);
    unsigned short* Xb   = (unsigned short*)alloc((size_t)NM * ND  * 2);
    unsigned short* QKVb = (unsigned short*)alloc((size_t)NM * TD  * 2);
    unsigned short* Ab   = (unsigned short*)alloc((size_t)NM * ND  * 2);
    float*          T1   = (float*)         alloc((size_t)NM * ND  * 4);
    float*          Nf   = (float*)         alloc((size_t)NM * ND  * 4);
    unsigned short* Nb   = (unsigned short*)alloc((size_t)NM * ND  * 2);
    unsigned short* Mb   = (unsigned short*)alloc((size_t)NM * NDF * 2);
    unsigned short* Wt   = (unsigned short*)alloc((size_t)NDF * ND * 2);  // max N*K
    (void)ws_size; (void)in_sizes; (void)n_in;

    embed_kernel<<<NM, 256, 0, stream>>>(tokens, we, X, Xb);

    for (int l = 0; l < NL; ++l) {
        const float* wq = wqkv  + (size_t)l * ND * TD;
        const float* bq = bqkv  + (size_t)l * TD;
        const float* wp = wproj + (size_t)l * ND * ND;
        const float* bp = bproj + (size_t)l * ND;
        const float* wf = wfc   + (size_t)l * ND * NDF;
        const float* bf = bfc   + (size_t)l * NDF;
        const float* wr = wpr   + (size_t)l * NDF * ND;
        const float* br = bpr   + (size_t)l * ND;

        // QKV projection: [4096,768] x [768,2304] -> bf16 QKV
        wtrans_kernel<<<dim3(TD / 64, ND / 64), 256, 0, stream>>>(wq, Wt, ND, TD);
        gemm_bf16_kernel<<<dim3(TD / 128, NM / 128), 256, 0, stream>>>(
            Xb, Wt, bq, nullptr, QKVb, NM, TD, ND, 0);
        // attention -> Ab (bf16)
        attn_kernel<<<dim3(NS / 128, NB * NH), 256, 0, stream>>>(QKVb, Ab);
        // output projection: [4096,768] x [768,768]
        wtrans_kernel<<<dim3(ND / 64, ND / 64), 256, 0, stream>>>(wp, Wt, ND, ND);
        gemm_bf16_kernel<<<dim3(ND / 128, NM / 128), 256, 0, stream>>>(
            Ab, Wt, bp, T1, nullptr, NM, ND, ND, 0);
        // n = LN(x + attn)
        add_ln_kernel<<<NM, 256, 0, stream>>>(X, T1, g1 + l * ND, b1 + l * ND, Nf, Nb);
        // FFN up + GELU: [4096,768] x [768,3072] -> bf16
        wtrans_kernel<<<dim3(NDF / 64, ND / 64), 256, 0, stream>>>(wf, Wt, ND, NDF);
        gemm_bf16_kernel<<<dim3(NDF / 128, NM / 128), 256, 0, stream>>>(
            Nb, Wt, bf, nullptr, Mb, NM, NDF, ND, 1);
        // FFN down: [4096,3072] x [3072,768]
        wtrans_kernel<<<dim3(ND / 64, NDF / 64), 256, 0, stream>>>(wr, Wt, NDF, ND);
        gemm_bf16_kernel<<<dim3(ND / 128, NM / 128), 256, 0, stream>>>(
            Mb, Wt, br, T1, nullptr, NM, ND, NDF, 0);
        // x = LN(n + mlp)
        add_ln_kernel<<<NM, 256, 0, stream>>>(Nf, T1, g2 + l * ND, b2 + l * ND, X, Xb);
    }

    hipMemcpyAsync(d_out, X, (size_t)NM * ND * sizeof(float),
                   hipMemcpyDeviceToDevice, stream);
}
